// NeuralNet_cnn_44315472560166
// MI455X (gfx1250) — matrix operand order repaired and checked by reading, not compiled, not run on hardware
//
#include <hip/hip_runtime.h>
#include <hip/hip_bf16.h>
#include <hip/hip_fp16.h>

typedef _Float16 v16h __attribute__((ext_vector_type(16)));
typedef _Float16 v8h  __attribute__((ext_vector_type(8)));
typedef float    v8f  __attribute__((ext_vector_type(8)));

#define NWAVES       4
#define GEMM_BLOCKS  512
#define STATS_BLOCKS 256
#define BN_EPS       1e-5f

// ---------------------------------------------------------------------------
// Fragment-native tiled activation layout (f16):
//   group g = 16 samples, K-tile kt = 32 features; tile = 512 halfs (1KB),
//   contiguous at (g*3 + kt)*512. Element (lane L, half j) of a tile holds
//   feature k = kt*32 + 16*(L>=16) + j, sample s = g*16 + (L&15)  -- exactly
//   the contiguous-k (k = 16h+s) B order this GEMM uses (A matches), so a wave's B fragment is
//   two contiguous global_load_b128 per lane; a sample's 16-feature run is a
//   contiguous 32B chunk for the scalar-side kernels.
// ---------------------------------------------------------------------------
__device__ __forceinline__ size_t tix(int f, int s) {
    int kt = f >> 5, r = f & 31, up = r >> 4, j = r & 15;
    return (size_t)(s >> 4) * 1536 + kt * 512 + (((s & 15) + (up << 4)) << 4) + j;
}

// ---------------------------------------------------------------------------
// Input prep: x [B,3,3,3] f32 -> tiled f16 H0 (features 27..95 zeroed).
// ---------------------------------------------------------------------------
__global__ void prep_input(const float* __restrict__ x, _Float16* __restrict__ H, int B) {
    int f = blockIdx.y;
    int s = blockIdx.x * blockDim.x + threadIdx.x;
    if (s >= B) return;
    float v = (f < 27) ? x[(size_t)s * 27 + f] : 0.f;
    H[tix(f, s)] = (_Float16)v;
}

// ---------------------------------------------------------------------------
// Conv layer as WMMA GEMM:  Y = A(96x96) * H; A built in LDS from conv weights
// (implicit GEMM of 3x3 SAME conv on 3x3 grid; conv bias cancels in BN).
// Software-pipelined: next group's B fragments load while current computes.
// ---------------------------------------------------------------------------
__global__ __launch_bounds__(NWAVES * 32)
void conv_gemm_wmma(const _Float16* __restrict__ Hin, _Float16* __restrict__ Yout,
                    const float* __restrict__ W, int IC, int B) {
    __shared__ _Float16 Alds[96 * 96];
    int tid = threadIdx.x;

    for (int idx = tid; idx < 96 * 96; idx += blockDim.x) {
        int m = idx / 96, k = idx % 96;
        float val = 0.f;
        if (m < 90 && k < IC * 9) {
            int oc = m / 9, op = m % 9, ic = k / 9, ip = k % 9;
            int oy = op / 3, ox = op % 3, iy = ip / 3, ix = ip % 3;
            int ky = iy - oy + 1, kx = ix - ox + 1;
            if ((unsigned)ky < 3u && (unsigned)kx < 3u)
                val = W[((oc * IC + ic) * 3 + ky) * 3 + kx];
        }
        Alds[idx] = (_Float16)val;
    }
    __syncthreads();

    int wave = tid >> 5, lane = tid & 31;
    int n  = lane & 15;
    int hi = lane >> 4;
    int ngroups = B >> 4;

    // A fragments (loop-invariant): lane m = mt*16+n, kbase = hi*16,
    // halves [0..15] = K kbase..kbase+15
    v16h afrag[6][3];
#pragma unroll
    for (int mt = 0; mt < 6; ++mt)
#pragma unroll
        for (int kt = 0; kt < 3; ++kt) {
            const v8h* ap = (const v8h*)&Alds[(mt * 16 + n) * 96 + kt * 32 + hi * 16];
            afrag[mt][kt] = __builtin_shufflevector(ap[0], ap[1],
                0, 1, 2, 3, 4, 5, 6, 7, 8, 9, 10, 11, 12, 13, 14, 15);
        }

    auto loadB = [&](v16h* bf, int gg) {
        const v8h* bt = (const v8h*)(Hin + (size_t)gg * 1536) + lane * 2;
#pragma unroll
        for (int kt = 0; kt < 3; ++kt)
            bf[kt] = __builtin_shufflevector(bt[kt * 64], bt[kt * 64 + 1],
                0, 1, 2, 3, 4, 5, 6, 7, 8, 9, 10, 11, 12, 13, 14, 15);
    };
    auto compute = [&](int gg, const v16h* bf) {
        _Float16* yt = Yout + (size_t)gg * 1536;
#pragma unroll
        for (int mt = 0; mt < 6; ++mt) {
            v8f acc = {0.f, 0.f, 0.f, 0.f, 0.f, 0.f, 0.f, 0.f};
#pragma unroll
            for (int kt = 0; kt < 3; ++kt)
                acc = __builtin_amdgcn_wmma_f32_16x16x32_f16(
                    false, afrag[mt][kt], false, bf[kt], (short)0, acc, false, false);
            // D rows mt*16+hi*8+r -> dest tile kt=mt>>1, dest lane n+16*(mt&1),
            // halves hi*8..hi*8+7: one packed 16B store.
            v8h hv;
#pragma unroll
            for (int r = 0; r < 8; ++r) hv[r] = (_Float16)acc[r];
            *(v8h*)(yt + ((mt >> 1) * 512 + ((n + ((mt & 1) << 4)) << 4) + hi * 8)) = hv;
        }
    };

    int gstep = gridDim.x * NWAVES;
    int g = blockIdx.x * NWAVES + wave;
    if (g >= ngroups) return;

    v16h b0[3], b1[3];
    loadB(b0, g);
    while (true) {
        int g1 = g + gstep;
        if (g1 >= ngroups) { compute(g, b0); break; }
        loadB(b1, g1);                                   // in flight during compute(g)
        int g2 = g1 + gstep;
        if (g2 < ngroups)
            __builtin_prefetch((const char*)(Hin + (size_t)g2 * 1536) + lane * 192, 0, 1);
        compute(g, b0);
        if (g2 >= ngroups) { compute(g1, b1); break; }
        loadB(b0, g2);                                   // in flight during compute(g1)
        if (g2 + gstep < ngroups)
            __builtin_prefetch((const char*)(Hin + (size_t)(g2 + gstep) * 1536) + lane * 192, 0, 1);
        compute(g1, b1);
        g = g2;
    }
}

// ---------------------------------------------------------------------------
// Per-channel sum/sumsq partials over tiled Y. One slot = 16 contiguous halfs
// (one sample, one 16-feature run); 6 (kt,up) combos per sample => feature
// k = combo*16+j is compile-time, so channel k/9 indexes registers constantly.
// Deterministic fixed-order tree reduction. part: [block][20]={sum[10],sq[10]}
// ---------------------------------------------------------------------------
__global__ void stats_reduce(const _Float16* __restrict__ Y, float* __restrict__ part, int B) {
    __shared__ float red[256];
    int tid = threadIdx.x;
    float sum[10], sq[10];
#pragma unroll
    for (int c = 0; c < 10; ++c) { sum[c] = 0.f; sq[c] = 0.f; }

    for (int s = blockIdx.x * blockDim.x + tid; s < B; s += gridDim.x * blockDim.x) {
        const v8h* p = (const v8h*)(Y + (size_t)(s >> 4) * 1536 + ((s & 15) << 4));
#pragma unroll
        for (int combo = 0; combo < 6; ++combo) {        // kt = combo>>1, up = combo&1
            v8h a = p[combo * 32];
            v8h b = p[combo * 32 + 1];
#pragma unroll
            for (int j = 0; j < 16; ++j) {
                int k = combo * 16 + j;                  // compile-time feature id
                if (k < 90) {
                    int c = k / 9;                       // compile-time channel
                    float v = (float)(j < 8 ? a[j] : b[j - 8]);
                    sum[c] += v; sq[c] += v * v;
                }
            }
        }
    }
#pragma unroll
    for (int c = 0; c < 10; ++c) {
        red[tid] = sum[c]; __syncthreads();
        for (int o = 128; o > 0; o >>= 1) { if (tid < o) red[tid] += red[tid + o]; __syncthreads(); }
        if (tid == 0) part[blockIdx.x * 20 + c] = red[0];
        __syncthreads();
        red[tid] = sq[c]; __syncthreads();
        for (int o = 128; o > 0; o >>= 1) { if (tid < o) red[tid] += red[tid + o]; __syncthreads(); }
        if (tid == 0) part[blockIdx.x * 20 + 10 + c] = red[0];
        __syncthreads();
    }
}

// ---------------------------------------------------------------------------
// Combine partials -> scale/shift (serial fixed-order => deterministic).
// ---------------------------------------------------------------------------
__global__ void bn_finalize(const float* __restrict__ part, int nblocks, int stride, int sqoff,
                            const float* __restrict__ g, const float* __restrict__ bt,
                            float* __restrict__ out, int nch, float invcount) {
    int c = threadIdx.x;
    if (c >= nch) return;
    float s = 0.f, q = 0.f;
    for (int b = 0; b < nblocks; ++b) {
        s += part[b * stride + c];
        q += part[b * stride + sqoff + c];
    }
    float mean = s * invcount;
    float var  = q * invcount - mean * mean;
    float sc   = g[c] * rsqrtf(var + BN_EPS);
    out[c]      = sc;
    out[10 + c] = bt[c] - mean * sc;
}

// ---------------------------------------------------------------------------
// In-place BN affine + optional residual + ReLU, vectorized: one thread = one
// 16-half contiguous run. Feature of half j is (q & 0x70) + j, q = slot % 96.
// Padding features 90..95 get scale=shift=0 -> stay exactly zero.
// ---------------------------------------------------------------------------
__global__ void bn_apply(_Float16* __restrict__ Y, const _Float16* __restrict__ Res,
                         const float* __restrict__ st, int B) {
    __shared__ float2 tbl[96];
    int tid = threadIdx.x;
    if (tid < 96) {
        float2 v = {0.f, 0.f};
        if (tid < 90) { int c = tid / 9; v.x = st[c]; v.y = st[10 + c]; }
        tbl[tid] = v;
    }
    __syncthreads();

    int nslots = (B >> 4) * 96;
    int t = blockIdx.x * blockDim.x + tid;
    if (t >= nslots) return;
    int q  = t % 96;
    int fb = q & 0x70;

    const v8h* yp = (const v8h*)Y + (size_t)t * 2;
    v8h lo = yp[0], hi8 = yp[1];
    v8h rlo = {}, rhi = {};
    if (Res) {
        const v8h* rp = (const v8h*)Res + (size_t)t * 2;
        rlo = rp[0]; rhi = rp[1];
    }
    v8h olo, ohi;
#pragma unroll
    for (int j = 0; j < 8; ++j) {
        float2 s0 = tbl[fb + j];
        float2 s1 = tbl[fb + 8 + j];
        float a = s0.x * (float)lo[j]  + s0.y + (float)rlo[j];
        float b = s1.x * (float)hi8[j] + s1.y + (float)rhi[j];
        olo[j] = (_Float16)fmaxf(a, 0.f);
        ohi[j] = (_Float16)fmaxf(b, 0.f);
    }
    ((v8h*)Y)[(size_t)t * 2]     = olo;
    ((v8h*)Y)[(size_t)t * 2 + 1] = ohi;
}

// ---------------------------------------------------------------------------
// Heads: 10->1 convs as 9x90 mat-vec per sample (vectorized b128 activation
// loads) + deterministic stats partials. hpart: [block][4]={ps,pq,vs,vq}
// ---------------------------------------------------------------------------
__global__ void head_gemm(const _Float16* __restrict__ H, float* __restrict__ p_pre,
                          float* __restrict__ v_pre, const float* __restrict__ Wp,
                          const float* __restrict__ Wv, float* __restrict__ hpart, int B) {
    __shared__ float Ap[9 * 90];
    __shared__ float Av[9 * 90];
    __shared__ float red[256];
    int tid = threadIdx.x;

    for (int idx = tid; idx < 810; idx += blockDim.x) {
        int j = idx / 90, f = idx % 90;
        int ic = f / 9, ip = f % 9;
        int oy = j / 3, ox = j % 3, iy = ip / 3, ix = ip % 3;
        int ky = iy - oy + 1, kx = ix - ox + 1;
        float vp = 0.f, vv = 0.f;
        if ((unsigned)ky < 3u && (unsigned)kx < 3u) {
            vp = Wp[(ic * 3 + ky) * 3 + kx];
            vv = Wv[(ic * 3 + ky) * 3 + kx];
        }
        Ap[idx] = vp; Av[idx] = vv;
    }
    __syncthreads();

    int s = blockIdx.x * blockDim.x + tid;
    float accp[9], accv[9];
#pragma unroll
    for (int j = 0; j < 9; ++j) { accp[j] = 0.f; accv[j] = 0.f; }

    float ps = 0.f, pq = 0.f, vs = 0.f, vq = 0.f;
    if (s < B) {
        const v8h* p = (const v8h*)(H + (size_t)(s >> 4) * 1536 + ((s & 15) << 4));
#pragma unroll
        for (int combo = 0; combo < 6; ++combo) {
            v8h a = p[combo * 32];
            v8h b = p[combo * 32 + 1];
#pragma unroll
            for (int j = 0; j < 16; ++j) {
                int k = combo * 16 + j;                  // compile-time feature id
                if (k < 90) {
                    float hv = (float)(j < 8 ? a[j] : b[j - 8]);
#pragma unroll
                    for (int j9 = 0; j9 < 9; ++j9) {
                        accp[j9] += Ap[j9 * 90 + k] * hv;
                        accv[j9] += Av[j9 * 90 + k] * hv;
                    }
                }
            }
        }
#pragma unroll
        for (int j = 0; j < 9; ++j) {
            p_pre[(size_t)j * B + s] = accp[j];
            v_pre[(size_t)j * B + s] = accv[j];
            ps += accp[j]; pq += accp[j] * accp[j];
            vs += accv[j]; vq += accv[j] * accv[j];
        }
    }
    float vals[4] = {ps, pq, vs, vq};
#pragma unroll
    for (int v = 0; v < 4; ++v) {
        __syncthreads();
        red[tid] = vals[v]; __syncthreads();
        for (int o = 128; o > 0; o >>= 1) { if (tid < o) red[tid] += red[tid + o]; __syncthreads(); }
        if (tid == 0) hpart[blockIdx.x * 4 + v] = red[0];
    }
}

// ---------------------------------------------------------------------------
// Final: policy BN->hardtanh, value BN->relu->fc(9x9)->relu->fc(9)->tanh.
// d_out = [out_p (B*9, sample-major)] ++ [out_v (B)]
// ---------------------------------------------------------------------------
__global__ void final_head(const float* __restrict__ p_pre, const float* __restrict__ v_pre,
                           const float* __restrict__ stp, const float* __restrict__ stv,
                           const float* __restrict__ fc2w, const float* __restrict__ fc2b,
                           const float* __restrict__ fc3w, const float* __restrict__ fc3b,
                           float* __restrict__ out, int B) {
    int s = blockIdx.x * blockDim.x + threadIdx.x;
    if (s >= B) return;
    float sp = stp[0], bp = stp[10];
    float sv = stv[0], bv = stv[10];
    float v1[9];
#pragma unroll
    for (int j = 0; j < 9; ++j) {
        float pv = sp * p_pre[(size_t)j * B + s] + bp;
        out[(size_t)s * 9 + j] = fminf(fmaxf(pv, -1.f), 1.f);
        v1[j] = fmaxf(sv * v_pre[(size_t)j * B + s] + bv, 0.f);
    }
    float v2[9];
#pragma unroll
    for (int j = 0; j < 9; ++j) {
        float a = fc2b[j];
#pragma unroll
        for (int k = 0; k < 9; ++k) a += fc2w[j * 9 + k] * v1[k];
        v2[j] = fmaxf(a, 0.f);
    }
    float a = fc3b[0];
#pragma unroll
    for (int k = 0; k < 9; ++k) a += fc3w[k] * v2[k];
    out[(size_t)9 * B + s] = tanhf(a);
}

// ---------------------------------------------------------------------------
extern "C" void kernel_launch(void* const* d_in, const int* in_sizes, int n_in,
                              void* d_out, int out_size, void* d_ws, size_t ws_size,
                              hipStream_t stream) {
    (void)n_in; (void)out_size; (void)ws_size;
    const int B = in_sizes[0] / 27;
    const float invc = 1.0f / (9.0f * (float)B);

    const float* x = (const float*)d_in[0];
    // conv{i}_w at 1+2(i-1); bn{i}_g at 15+2(i-1), bn{i}_b at 16+2(i-1)
    const float* convp_w = (const float*)d_in[29];
    const float* bnp_g   = (const float*)d_in[31];
    const float* bnp_b   = (const float*)d_in[32];
    const float* convv_w = (const float*)d_in[33];
    const float* bnv_g   = (const float*)d_in[35];
    const float* bnv_b   = (const float*)d_in[36];
    const float* fc2w    = (const float*)d_in[37];
    const float* fc2b    = (const float*)d_in[38];
    const float* fc3w    = (const float*)d_in[39];
    const float* fc3b    = (const float*)d_in[40];

    // Workspace layout
    _Float16* P0 = (_Float16*)d_ws;
    _Float16* P1 = P0 + (size_t)96 * B;
    _Float16* P2 = P1 + (size_t)96 * B;
    float* p_pre = (float*)(P2 + (size_t)96 * B);
    float* v_pre = p_pre + (size_t)9 * B;
    float* stats = v_pre + (size_t)9 * B;       // 9 slots x {scale[10], shift[10]}
    float* part  = stats + 9 * 20;              // STATS_BLOCKS x 20
    float* hpart = part + STATS_BLOCKS * 20;    // (B/256) x 4

    int nsb = (B + 255) / 256;
    int nbn = ((B >> 4) * 96 + 255) / 256;      // bn_apply slot blocks

    prep_input<<<dim3(nsb, 96), 256, 0, stream>>>(x, P0, B);

    auto layer = [&](const _Float16* in, _Float16* out, const _Float16* res, int li, int IC) {
        const float* W  = (const float*)d_in[1 + 2 * (li - 1)];
        const float* g  = (const float*)d_in[15 + 2 * (li - 1)];
        const float* bt = (const float*)d_in[16 + 2 * (li - 1)];
        float* st = stats + (li - 1) * 20;
        conv_gemm_wmma<<<GEMM_BLOCKS, NWAVES * 32, 0, stream>>>(in, out, W, IC, B);
        stats_reduce<<<STATS_BLOCKS, 256, 0, stream>>>(out, part, B);
        bn_finalize<<<1, 32, 0, stream>>>(part, STATS_BLOCKS, 20, 10, g, bt, st, 10, invc);
        bn_apply<<<nbn, 256, 0, stream>>>(out, res, st, B);
    };

    layer(P0, P1, nullptr, 1, 3);   // out1 = P1
    layer(P1, P2, nullptr, 2, 10);  // out2 = P2
    layer(P2, P0, P1,      3, 10);  // out3 = P0 (res out1)
    layer(P0, P1, nullptr, 4, 10);  // out4 = P1
    layer(P1, P2, P0,      5, 10);  // out5 = P2 (res out3)
    layer(P2, P0, nullptr, 6, 10);  // out6 = P0
    layer(P0, P1, P2,      7, 10);  // out7 = P1 (res out5)

    float* stp = stats + 7 * 20;
    float* stv = stats + 8 * 20;
    head_gemm<<<nsb, 256, 0, stream>>>(P1, p_pre, v_pre, convp_w, convv_w, hpart, B);
    bn_finalize<<<1, 32, 0, stream>>>(hpart + 0, nsb, 4, 1, bnp_g, bnp_b, stp, 1, invc);
    bn_finalize<<<1, 32, 0, stream>>>(hpart + 2, nsb, 4, 1, bnv_g, bnv_b, stv, 1, invc);
    final_head<<<nsb, 256, 0, stream>>>(p_pre, v_pre, stp, stv, fc2w, fc2b, fc3w, fc3b,
                                        (float*)d_out, B);
}
